// NeuralODE_18056042513170
// MI455X (gfx1250) — matrix operand order repaired and checked by reading, not compiled, not run on hardware
//
#include <hip/hip_runtime.h>
#include <hip/hip_bf16.h>
#include <math.h>

// Problem constants (from reference)
#define BS 512
#define DD 1024
#define HH 4096
#define TT 64

typedef __attribute__((ext_vector_type(16))) __bf16 v16bf;
typedef __attribute__((ext_vector_type(8)))  __bf16 v8bf;
typedef __attribute__((ext_vector_type(8)))  float  v8f;

// 3-term split-bf16 product accumulate:  C += Ahi*Bhi + Ahi*Blo + Alo*Bhi
static __device__ __forceinline__ v8f mma3(v16bf ahi, v16bf alo,
                                           v16bf bhi, v16bf blo, v8f c) {
    c = __builtin_amdgcn_wmma_f32_16x16x32_bf16(false, ahi, false, bhi,
                                                (short)0, c, false, false);
    c = __builtin_amdgcn_wmma_f32_16x16x32_bf16(false, ahi, false, blo,
                                                (short)0, c, false, false);
    c = __builtin_amdgcn_wmma_f32_16x16x32_bf16(false, alo, false, bhi,
                                                (short)0, c, false, false);
    return c;
}

// Load one 16x32 bf16 A-fragment from a row-major (stride `ld`) bf16 matrix.
// Layout: lane = row (lane&15); lanes<16 hold K=kb..kb+15,
// lanes>=16 hold K=kb+16..kb+31.
static __device__ __forceinline__ v16bf load_afrag(const __bf16* __restrict__ M,
                                                   int m, int k, int half,
                                                   int ld) {
    const __bf16* p = M + (size_t)m * ld + k + half * 16;
    v8bf a0 = *(const v8bf*)(p);
    v8bf a1 = *(const v8bf*)(p + 8);
    return __builtin_shufflevector(a0, a1, 0,1,2,3,4,5,6,7,8,9,10,11,12,13,14,15);
}

// ---------------------------------------------------------------------------
// Split a fp32 matrix into (hi, lo) bf16 while transposing:  WT[n][k] = W[k][n]
// so B-fragment lanes read contiguous 32B runs along K.
// ---------------------------------------------------------------------------
__global__ void node_split_transpose(const float* __restrict__ W,
                                     __bf16* __restrict__ hi,
                                     __bf16* __restrict__ lo,
                                     int rows, int cols) {
    size_t idx = (size_t)blockIdx.x * blockDim.x + threadIdx.x;
    size_t n_el = (size_t)rows * cols;
    if (idx >= n_el) return;
    int k = (int)(idx / cols);
    int n = (int)(idx % cols);
    float v = W[idx];
    __bf16 h = (__bf16)v;
    size_t o = (size_t)n * rows + k;
    hi[o] = h;
    lo[o] = (__bf16)(v - (float)h);
}

// z0 -> trajectory slice 0, fp32 ping buffer, and split bf16 hi/lo
__global__ void node_copy_z0(const float* __restrict__ z0,
                             float* __restrict__ traj0,
                             float* __restrict__ zA,
                             __bf16* __restrict__ zhi,
                             __bf16* __restrict__ zlo, int n) {
    int i = blockIdx.x * blockDim.x + threadIdx.x;
    if (i < n) {
        float v = z0[i];
        traj0[i] = v;
        zA[i] = v;
        __bf16 h = (__bf16)v;
        zhi[i] = h;
        zlo[i] = (__bf16)(v - (float)h);
    }
}

// ---------------------------------------------------------------------------
// GEMM1:  Hmat = tanh( Z[512x1024] * W1[1024x4096] + b1 )
// Z arrives pre-split (Zhi/Zlo bf16). Wave tile 32x32 (4 accumulators,
// 12 WMMAs per k-step). WG = 4 waves covering 32(M) x 128(N).
// grid = (512/32) * (4096/128) = 512 WGs, block = 128.
// Output stored split (hi/lo bf16) for GEMM2.
// ---------------------------------------------------------------------------
__global__ void __launch_bounds__(128)
node_gemm1_tanh(const __bf16* __restrict__ Zhi,
                const __bf16* __restrict__ Zlo,
                const __bf16* __restrict__ W1T_hi,
                const __bf16* __restrict__ W1T_lo,
                const float* __restrict__ b1,
                __bf16* __restrict__ Hhi,
                __bf16* __restrict__ Hlo) {
    const int lane = threadIdx.x & 31;
    const int wave = threadIdx.x >> 5;
    const int wgM  = blockIdx.x & 15;       // 16 M-tiles of 32
    const int wgN  = blockIdx.x >> 4;       // 32 N-tiles of 128
    const int m0   = wgM * 32;
    const int n0   = wgN * 128 + wave * 32;

    const int half = lane >> 4;
    const int lmod = lane & 15;

    v8f acc00 = {}, acc01 = {}, acc10 = {}, acc11 = {};

    for (int k = 0; k < DD; k += 32) {
        // Two 16-row A fragments (hi/lo) from pre-split Z
        v16bf a0hi = load_afrag(Zhi, m0 + lmod,      k, half, DD);
        v16bf a0lo = load_afrag(Zlo, m0 + lmod,      k, half, DD);
        v16bf a1hi = load_afrag(Zhi, m0 + 16 + lmod, k, half, DD);
        v16bf a1lo = load_afrag(Zlo, m0 + 16 + lmod, k, half, DD);

        const int kb2 = k + half * 16;
        const int c0  = n0 + lmod;
        const int c1  = n0 + 16 + lmod;
        const __bf16* bh0 = W1T_hi + (size_t)c0 * DD + kb2;
        const __bf16* bl0 = W1T_lo + (size_t)c0 * DD + kb2;
        const __bf16* bh1 = W1T_hi + (size_t)c1 * DD + kb2;
        const __bf16* bl1 = W1T_lo + (size_t)c1 * DD + kb2;
        __builtin_prefetch(bh0 + 32, 0, 0);
        __builtin_prefetch(bh1 + 32, 0, 0);
        v16bf b0hi = *(const v16bf*)bh0;
        v16bf b0lo = *(const v16bf*)bl0;
        v16bf b1hi = *(const v16bf*)bh1;
        v16bf b1lo = *(const v16bf*)bl1;

        acc00 = mma3(a0hi, a0lo, b0hi, b0lo, acc00);
        acc01 = mma3(a0hi, a0lo, b1hi, b1lo, acc01);
        acc10 = mma3(a1hi, a1lo, b0hi, b0lo, acc10);
        acc11 = mma3(a1hi, a1lo, b1hi, b1lo, acc11);
    }

    // Epilogue: bias + tanh, split-store bf16 hi/lo
    const int col0 = n0 + lmod;
    const int col1 = n0 + 16 + lmod;
    const float bias0 = b1[col0];
    const float bias1 = b1[col1];
#pragma unroll
    for (int r = 0; r < 8; ++r) {
        const int row0 = m0 + r + 8 * half;
        const int row1 = row0 + 16;
        float v00 = tanhf(acc00[r] + bias0);
        float v01 = tanhf(acc01[r] + bias1);
        float v10 = tanhf(acc10[r] + bias0);
        float v11 = tanhf(acc11[r] + bias1);
        __bf16 h00 = (__bf16)v00, h01 = (__bf16)v01;
        __bf16 h10 = (__bf16)v10, h11 = (__bf16)v11;
        size_t o00 = (size_t)row0 * HH + col0;
        size_t o01 = (size_t)row0 * HH + col1;
        size_t o10 = (size_t)row1 * HH + col0;
        size_t o11 = (size_t)row1 * HH + col1;
        Hhi[o00] = h00; Hlo[o00] = (__bf16)(v00 - (float)h00);
        Hhi[o01] = h01; Hlo[o01] = (__bf16)(v01 - (float)h01);
        Hhi[o10] = h10; Hlo[o10] = (__bf16)(v10 - (float)h10);
        Hhi[o11] = h11; Hlo[o11] = (__bf16)(v11 - (float)h11);
    }
}

// ---------------------------------------------------------------------------
// GEMM2:  Znext = Zcur + h * ( Hmat[512x4096] * W2[4096x1024] + b2 )
// Wave tile 32x32; WG = 4 waves covering 32(M) x 128(N).
// grid = (512/32) * (1024/128) = 128 WGs, block = 128.
// Epilogue also emits the split bf16 copy of znext (GEMM1's next input) and,
// on the interval's 4th substep, the trajectory slice.
// ---------------------------------------------------------------------------
__global__ void __launch_bounds__(128)
node_gemm2_euler(const __bf16* __restrict__ Hhi,
                 const __bf16* __restrict__ Hlo,
                 const __bf16* __restrict__ W2T_hi,
                 const __bf16* __restrict__ W2T_lo,
                 const float* __restrict__ b2,
                 const float* __restrict__ tvec,
                 int interval,
                 const float* __restrict__ zcur,
                 float* __restrict__ znext,
                 __bf16* __restrict__ znext_hi,
                 __bf16* __restrict__ znext_lo,
                 float* __restrict__ traj,
                 int write_traj) {
    const int lane = threadIdx.x & 31;
    const int wave = threadIdx.x >> 5;
    const int wgM  = blockIdx.x & 15;       // 16 M-tiles of 32
    const int wgN  = blockIdx.x >> 4;       // 8 N-tiles of 128
    const int m0   = wgM * 32;
    const int n0   = wgN * 128 + wave * 32;

    const int half = lane >> 4;
    const int lmod = lane & 15;

    v8f acc00 = {}, acc01 = {}, acc10 = {}, acc11 = {};

    for (int k = 0; k < HH; k += 32) {
        v16bf a0hi = load_afrag(Hhi, m0 + lmod,      k, half, HH);
        v16bf a0lo = load_afrag(Hlo, m0 + lmod,      k, half, HH);
        v16bf a1hi = load_afrag(Hhi, m0 + 16 + lmod, k, half, HH);
        v16bf a1lo = load_afrag(Hlo, m0 + 16 + lmod, k, half, HH);

        const int kb2 = k + half * 16;
        const int c0  = n0 + lmod;
        const int c1  = n0 + 16 + lmod;
        const __bf16* bh0 = W2T_hi + (size_t)c0 * HH + kb2;
        const __bf16* bl0 = W2T_lo + (size_t)c0 * HH + kb2;
        const __bf16* bh1 = W2T_hi + (size_t)c1 * HH + kb2;
        const __bf16* bl1 = W2T_lo + (size_t)c1 * HH + kb2;
        __builtin_prefetch(bh0 + 32, 0, 0);
        __builtin_prefetch(bh1 + 32, 0, 0);
        v16bf b0hi = *(const v16bf*)bh0;
        v16bf b0lo = *(const v16bf*)bl0;
        v16bf b1hi = *(const v16bf*)bh1;
        v16bf b1lo = *(const v16bf*)bl1;

        acc00 = mma3(a0hi, a0lo, b0hi, b0lo, acc00);
        acc01 = mma3(a0hi, a0lo, b1hi, b1lo, acc01);
        acc10 = mma3(a1hi, a1lo, b0hi, b0lo, acc10);
        acc11 = mma3(a1hi, a1lo, b1hi, b1lo, acc11);
    }

    // Euler update: h = |t[i+1]-t[i]| / N_INNER
    const float dt = fabsf(tvec[interval + 1] - tvec[interval]);
    const float hstep = 0.25f * dt;

    const int col0 = n0 + lmod;
    const int col1 = n0 + 16 + lmod;
    const float bias0 = b2[col0];
    const float bias1 = b2[col1];
#pragma unroll
    for (int r = 0; r < 8; ++r) {
        const int row0 = m0 + r + 8 * half;
        const int row1 = row0 + 16;
        const size_t o00 = (size_t)row0 * DD + col0;
        const size_t o01 = (size_t)row0 * DD + col1;
        const size_t o10 = (size_t)row1 * DD + col0;
        const size_t o11 = (size_t)row1 * DD + col1;
        float z00 = zcur[o00] + hstep * (acc00[r] + bias0);
        float z01 = zcur[o01] + hstep * (acc01[r] + bias1);
        float z10 = zcur[o10] + hstep * (acc10[r] + bias0);
        float z11 = zcur[o11] + hstep * (acc11[r] + bias1);
        znext[o00] = z00; znext[o01] = z01; znext[o10] = z10; znext[o11] = z11;
        __bf16 h00 = (__bf16)z00, h01 = (__bf16)z01;
        __bf16 h10 = (__bf16)z10, h11 = (__bf16)z11;
        znext_hi[o00] = h00; znext_lo[o00] = (__bf16)(z00 - (float)h00);
        znext_hi[o01] = h01; znext_lo[o01] = (__bf16)(z01 - (float)h01);
        znext_hi[o10] = h10; znext_lo[o10] = (__bf16)(z10 - (float)h10);
        znext_hi[o11] = h11; znext_lo[o11] = (__bf16)(z11 - (float)h11);
        if (write_traj) {
            traj[o00] = z00; traj[o01] = z01; traj[o10] = z10; traj[o11] = z11;
        }
    }
}

// ---------------------------------------------------------------------------
// Workspace layout (bytes from d_ws):
//   0  MB : W1T_hi [4096x1024] bf16 (8 MB)     8  MB : W1T_lo (8 MB)
//   16 MB : W2T_hi [1024x4096] bf16 (8 MB)     24 MB : W2T_lo (8 MB)
//   32 MB : H_hi   [512x4096]  bf16 (4 MB)     36 MB : H_lo   (4 MB)
//   40 MB : zA f32 (2 MB)   42 MB : zB f32 (2 MB)
//   44 MB : zA_hi (1 MB)  45 MB : zA_lo (1 MB)  46 MB : zB_hi (1 MB)  47 MB : zB_lo (1 MB)
// ---------------------------------------------------------------------------
extern "C" void kernel_launch(void* const* d_in, const int* in_sizes, int n_in,
                              void* d_out, int out_size, void* d_ws, size_t ws_size,
                              hipStream_t stream) {
    const float* z0 = (const float*)d_in[0];
    const float* tv = (const float*)d_in[1];
    const float* W1 = (const float*)d_in[2];
    const float* b1 = (const float*)d_in[3];
    const float* W2 = (const float*)d_in[4];
    const float* b2 = (const float*)d_in[5];
    float* out = (float*)d_out;

    char* ws = (char*)d_ws;
    const size_t MB = 1024 * 1024;
    __bf16* W1T_hi = (__bf16*)(ws + 0 * MB);
    __bf16* W1T_lo = (__bf16*)(ws + 8 * MB);
    __bf16* W2T_hi = (__bf16*)(ws + 16 * MB);
    __bf16* W2T_lo = (__bf16*)(ws + 24 * MB);
    __bf16* Hhi    = (__bf16*)(ws + 32 * MB);
    __bf16* Hlo    = (__bf16*)(ws + 36 * MB);
    float*  zA     = (float*) (ws + 40 * MB);
    float*  zB     = (float*) (ws + 42 * MB);
    __bf16* zAhi   = (__bf16*)(ws + 44 * MB);
    __bf16* zAlo   = (__bf16*)(ws + 45 * MB);
    __bf16* zBhi   = (__bf16*)(ws + 46 * MB);
    __bf16* zBlo   = (__bf16*)(ws + 47 * MB);

    // 1) Split+transpose weights into bf16 hi/lo.
    {
        const size_t n1 = (size_t)DD * HH;
        node_split_transpose<<<(unsigned)((n1 + 255) / 256), 256, 0, stream>>>(
            W1, W1T_hi, W1T_lo, DD, HH);
        const size_t n2 = (size_t)HH * DD;
        node_split_transpose<<<(unsigned)((n2 + 255) / 256), 256, 0, stream>>>(
            W2, W2T_hi, W2T_lo, HH, DD);
    }

    // 2) z0 -> trajectory[0], fp32 ping buffer, split bf16 copy.
    {
        const int n = BS * DD;
        node_copy_z0<<<(n + 255) / 256, 256, 0, stream>>>(z0, out, zA, zAhi, zAlo, n);
    }

    // 3) 63 intervals x 4 Euler substeps; ping-pong (f32 + split) z buffers.
    float*  zc  = zA;  float*  zn  = zB;
    __bf16* zch = zAhi; __bf16* zcl = zAlo;
    __bf16* znh = zBhi; __bf16* znl = zBlo;
    for (int interval = 0; interval < TT - 1; ++interval) {
        float* traj = out + (size_t)(interval + 1) * BS * DD;
        for (int sub = 0; sub < 4; ++sub) {
            node_gemm1_tanh<<<512, 128, 0, stream>>>(zch, zcl, W1T_hi, W1T_lo,
                                                     b1, Hhi, Hlo);
            node_gemm2_euler<<<128, 128, 0, stream>>>(Hhi, Hlo, W2T_hi, W2T_lo, b2,
                                                      tv, interval, zc, zn, znh, znl,
                                                      traj, (sub == 3) ? 1 : 0);
            float* tf = zc; zc = zn; zn = tf;
            __bf16* th = zch; zch = znh; znh = th;
            __bf16* tl = zcl; zcl = znl; znl = tl;
        }
    }
    (void)in_sizes; (void)n_in; (void)out_size; (void)ws_size;
}
